// SelfAttention_11029476016812
// MI455X (gfx1250) — matrix operand order repaired and checked by reading, not compiled, not run on hardware
//
#include <hip/hip_runtime.h>
#include <hip/hip_bf16.h>

typedef unsigned short u16;
typedef unsigned int   u32;
typedef __attribute__((ext_vector_type(16))) __bf16 v16bf;
typedef __attribute__((ext_vector_type(8)))  __bf16 v8bf;
typedef __attribute__((ext_vector_type(8)))  float  v8f;

#define CCH    64
#define HWD    4096
#define NBATCH 8

union Frag  { v16bf v; v8bf h[2]; };
union FragU { v16bf v; u32  u[8]; };

__device__ __forceinline__ u16 f2bf(float f) {
  return __builtin_bit_cast(u16, (__bf16)f);
}

__device__ __forceinline__ u32 pack2(float a, float b) {  // -> v_cvt_pk_bf16_f32
  union { u16 h[2]; u32 u; } t;
  t.h[0] = f2bf(a); t.h[1] = f2bf(b);
  return t.u;
}

__device__ __forceinline__ u32 swz_xor16_u(u32 x) {       // lane n <-> n^16 (SWAPX16)
  return (u32)__builtin_amdgcn_ds_swizzle((int)x, 0x401F);
}
__device__ __forceinline__ float swz_xor16_f(float x) {
  int i = __builtin_bit_cast(int, x);
  i = __builtin_amdgcn_ds_swizzle(i, 0x401F);
  return __builtin_bit_cast(float, i);
}

__device__ __forceinline__ v8f wmma_bf16(v16bf a, v16bf b, v8f c) {
  return __builtin_amdgcn_wmma_f32_16x16x32_bf16(false, a, false, b, (short)0, c, false, false);
}

__device__ __forceinline__ uint4 pack8(const float* p) {
  union { u16 u[8]; uint4 q; } t;
#pragma unroll
  for (int i = 0; i < 8; ++i) t.u[i] = f2bf(p[i]);
  return t.q;
}

// ---------------------------------------------------------------------------
// Pass 1: q/k/v = W x + b (1x1 conv as GEMM, bf16 WMMA, fp32 accumulate).
// q gets 1/sqrt(C) * log2(e) folded in so pass 2 can use raw v_exp_f32 (exp2).
// Outputs: qT,kT as [N][HW][C] bf16 (fragment-friendly), v as [N][C][HW].
// ---------------------------------------------------------------------------
__global__ __launch_bounds__(128) void proj_kernel(
    const float* __restrict__ xq, const float* __restrict__ xk, const float* __restrict__ xv,
    const float* __restrict__ Wq, const float* __restrict__ bq,
    const float* __restrict__ Wk, const float* __restrict__ bk,
    const float* __restrict__ Wv, const float* __restrict__ bv,
    u16* __restrict__ qT, u16* __restrict__ kT, u16* __restrict__ vO)
{
  const int p    = blockIdx.z;
  const int n    = blockIdx.y;
  const int pb   = blockIdx.x * 64;
  const int tid  = threadIdx.x;
  const int lane = tid & 31, wave = tid >> 5;
  const int h    = lane >> 4, ln = lane & 15;

  const float* X = (p == 0) ? xq : (p == 1) ? xk : xv;
  const float* W = (p == 0) ? Wq : (p == 1) ? Wk : Wv;
  const float* B = (p == 0) ? bq : (p == 1) ? bk : bv;

  __shared__ __align__(16) u16 ldsX[64][CCH + 8];   // [pixel][channel] transposed stage

  const float* Xn = X + (size_t)n * CCH * HWD;
#pragma unroll
  for (int j = 0; j < 8; ++j) {
    int idx = j * 128 + tid;
    int c = idx >> 4, ch = idx & 15;
    float4 x4 = *(const float4*)(Xn + (size_t)c * HWD + pb + ch * 4);
    ldsX[ch * 4 + 0][c] = f2bf(x4.x);
    ldsX[ch * 4 + 1][c] = f2bf(x4.y);
    ldsX[ch * 4 + 2][c] = f2bf(x4.z);
    ldsX[ch * 4 + 3][c] = f2bf(x4.w);
  }
  __syncthreads();

  const int pcol = wave * 16 + ln;
  Frag bfr[2];
#pragma unroll
  for (int s = 0; s < 2; ++s) {
    bfr[s].h[0] = *(const v8bf*)&ldsX[pcol][s * 32 + h * 8];
    bfr[s].h[1] = *(const v8bf*)&ldsX[pcol][s * 32 + h * 8 + 16];
  }

  // q: fold attention scale AND log2(e) so softmax can use exp2 directly.
  const float scale = (p == 0) ? (0.125f * 1.44269504088896340736f) : 1.0f;
#pragma unroll
  for (int ot = 0; ot < 4; ++ot) {
    v8f acc = {0.f,0.f,0.f,0.f,0.f,0.f,0.f,0.f};
#pragma unroll
    for (int s = 0; s < 2; ++s) {
      const float* wr = W + (ot * 16 + ln) * CCH + s * 32 + h * 8;
      float4 w0 = *(const float4*)(wr);
      float4 w1 = *(const float4*)(wr + 4);
      float4 w2 = *(const float4*)(wr + 16);
      float4 w3 = *(const float4*)(wr + 20);
      Frag a;
      a.h[0] = (v8bf){(__bf16)w0.x,(__bf16)w0.y,(__bf16)w0.z,(__bf16)w0.w,
                      (__bf16)w1.x,(__bf16)w1.y,(__bf16)w1.z,(__bf16)w1.w};
      a.h[1] = (v8bf){(__bf16)w2.x,(__bf16)w2.y,(__bf16)w2.z,(__bf16)w2.w,
                      (__bf16)w3.x,(__bf16)w3.y,(__bf16)w3.z,(__bf16)w3.w};
      acc = wmma_bf16(a.v, bfr[s].v, acc);
    }
    const float* br = B + ot * 16 + h * 8;
    float4 b0 = *(const float4*)(br);
    float4 b1 = *(const float4*)(br + 4);
    float bias[8] = {b0.x,b0.y,b0.z,b0.w,b1.x,b1.y,b1.z,b1.w};
    float y[8];
#pragma unroll
    for (int r = 0; r < 8; ++r) y[r] = (acc[r] + bias[r]) * scale;

    if (p < 2) {
      u16* dst = ((p == 0) ? qT : kT) +
                 ((size_t)n * HWD + pb + pcol) * CCH + ot * 16 + h * 8;
      *(uint4*)dst = pack8(y);
    } else {
#pragma unroll
      for (int r = 0; r < 8; ++r)
        vO[((size_t)n * CCH + ot * 16 + h * 8 + r) * HWD + pb + pcol] = f2bf(y[r]);
    }
  }
}

// ---------------------------------------------------------------------------
// Pass 2: fused flash attention. 4 waves/block, 16 queries/wave, 32 keys/iter,
// double-buffered LDS (one barrier per iteration), exp2-domain online softmax,
// P reshuffled D-layout -> B-layout with ds_swizzle (no LDS round-trip).
// ---------------------------------------------------------------------------
__global__ __launch_bounds__(128) void attn_kernel(
    const float* __restrict__ query, const u16* __restrict__ qT,
    const u16* __restrict__ kT, const u16* __restrict__ vP,
    float* __restrict__ out)
{
  const int tid  = threadIdx.x;
  const int lane = tid & 31, wave = tid >> 5;
  const int h    = lane >> 4, ln = lane & 15;
  const int n    = blockIdx.y;
  const int qb   = blockIdx.x * 64;
  const int qpix = qb + wave * 16 + ln;

  __shared__ __align__(16) u16 ldsK[2][32][CCH + 8];   // kT tile: [x][c]
  __shared__ __align__(16) u16 ldsV[2][CCH][32 + 8];   // v  tile: [c][x]

  const u16* qTn = qT + (size_t)n * HWD * CCH;
  const u16* kTn = kT + (size_t)n * HWD * CCH;
  const u16* vn  = vP + (size_t)n * CCH * HWD;

  // Q B-fragments (K=c, N=query), reused for all 128 key blocks.
  Frag qf[2];
#pragma unroll
  for (int s = 0; s < 2; ++s) {
    const u16* src = qTn + (size_t)qpix * CCH + s * 32 + h * 8;
    qf[s].h[0] = *(const v8bf*)(src);
    qf[s].h[1] = *(const v8bf*)(src + 16);
  }

  v8f acc[4] = {};
  float m = -1e30f, l = 0.f;

  uint4 stgK[2], stgV[2];
  auto loadTile = [&](int xb) {
#pragma unroll
    for (int j = 0; j < 2; ++j) {
      int idx = j * 128 + tid;
      stgK[j] = *(const uint4*)(kTn + (size_t)(xb + (idx >> 3)) * CCH + (idx & 7) * 8);
      stgV[j] = *(const uint4*)(vn + (size_t)(idx >> 2) * HWD + xb + (idx & 3) * 8);
    }
  };
  auto storeTile = [&](int buf) {
#pragma unroll
    for (int j = 0; j < 2; ++j) {
      int idx = j * 128 + tid;
      *(uint4*)&ldsK[buf][idx >> 3][(idx & 7) * 8] = stgK[j];
      *(uint4*)&ldsV[buf][idx >> 2][(idx & 3) * 8] = stgV[j];
    }
  };

  loadTile(0);
  storeTile(0);
  __syncthreads();

  const int NIT = HWD / 32;
  for (int it = 0; it < NIT; ++it) {
    const int buf = it & 1;
    if (it + 1 < NIT) loadTile((it + 1) * 32);   // issue next tile's global loads

    // S[x(32)][y(16)] = K^T q : 2 M-tiles x 2 K-steps of bf16 WMMA.
    v8f S[2];
#pragma unroll
    for (int t = 0; t < 2; ++t) {
      v8f s = {0.f,0.f,0.f,0.f,0.f,0.f,0.f,0.f};
#pragma unroll
      for (int ss = 0; ss < 2; ++ss) {
        Frag a;  // A (M=x, K=c): row x = t*16+ln, K chunks {h*8, h*8+16}
        a.h[0] = *(const v8bf*)&ldsK[buf][t * 16 + ln][ss * 32 + h * 8];
        a.h[1] = *(const v8bf*)&ldsK[buf][t * 16 + ln][ss * 32 + h * 8 + 16];
        s = wmma_bf16(a.v, qf[ss].v, s);
      }
      S[t] = s;
    }

    // Online softmax over key axis, base-2 domain (log2e folded into q).
    float cmax = S[0][0];
#pragma unroll
    for (int r = 1; r < 8; ++r) cmax = fmaxf(cmax, S[0][r]);
#pragma unroll
    for (int r = 0; r < 8; ++r) cmax = fmaxf(cmax, S[1][r]);
    cmax = fmaxf(cmax, swz_xor16_f(cmax));
    const float mn = fmaxf(m, cmax);
    float p0[8], p1[8], lsum = 0.f;
#pragma unroll
    for (int r = 0; r < 8; ++r) { p0[r] = __builtin_amdgcn_exp2f(S[0][r] - mn); lsum += p0[r]; }
#pragma unroll
    for (int r = 0; r < 8; ++r) { p1[r] = __builtin_amdgcn_exp2f(S[1][r] - mn); lsum += p1[r]; }
    lsum += swz_xor16_f(lsum);
    const float alpha = __builtin_amdgcn_exp2f(m - mn);
    l = l * alpha + lsum;
    m = mn;
#pragma unroll
    for (int ct = 0; ct < 4; ++ct)
#pragma unroll
      for (int r = 0; r < 8; ++r) acc[ct][r] *= alpha;

    // Reshuffle P: C/D layout -> B layout via SWAPX16 register exchange.
    // lane h=0 owns x=0..7 (p0) / x=16..23 (p1); h=1 owns x=8..15 / x=24..31.
    // B-frag: lane<16 wants x=0..15, lane>=16 wants x=16..31.
    u32 P0[4], P1[4];
#pragma unroll
    for (int i = 0; i < 4; ++i) {
      P0[i] = pack2(p0[2*i], p0[2*i+1]);
      P1[i] = pack2(p1[2*i], p1[2*i+1]);
    }
    FragU pb;
#pragma unroll
    for (int i = 0; i < 4; ++i) {
      const u32 s0 = swz_xor16_u(P0[i]);
      const u32 s1 = swz_xor16_u(P1[i]);
      pb.u[i]     = h ? s1 : P0[i];   // low  8 K-values
      pb.u[4 + i] = h ? P1[i] : s0;   // high 8 K-values
    }

    // O[c][y] += V[c][x] * P[x][y] : 4 channel M-tiles, K=32.
#pragma unroll
    for (int ct = 0; ct < 4; ++ct) {
      Frag a;
      a.h[0] = *(const v8bf*)&ldsV[buf][ct * 16 + ln][h * 16];
      a.h[1] = *(const v8bf*)&ldsV[buf][ct * 16 + ln][h * 16 + 8];
      acc[ct] = wmma_bf16(a.v, pb.v, acc[ct]);
    }

    if (it + 1 < NIT) storeTile(buf ^ 1);        // fill the other buffer
    __syncthreads();                             // single barrier per iteration
  }

  // Epilogue: normalize, add residual (query), store fp32.
  const float inv = 1.0f / l;
#pragma unroll
  for (int ct = 0; ct < 4; ++ct) {
#pragma unroll
    for (int r = 0; r < 8; ++r) {
      const int c = ct * 16 + h * 8 + r;
      const size_t idx = ((size_t)n * CCH + c) * HWD + qpix;
      out[idx] = query[idx] + acc[ct][r] * inv;
    }
  }
}

extern "C" void kernel_launch(void* const* d_in, const int* in_sizes, int n_in,
                              void* d_out, int out_size, void* d_ws, size_t ws_size,
                              hipStream_t stream) {
  const float* q  = (const float*)d_in[0];
  const float* k  = (const float*)d_in[1];
  const float* v  = (const float*)d_in[2];
  const float* Wq = (const float*)d_in[3];
  const float* bq = (const float*)d_in[4];
  const float* Wk = (const float*)d_in[5];
  const float* bk = (const float*)d_in[6];
  const float* Wv = (const float*)d_in[7];
  const float* bv = (const float*)d_in[8];
  float* out = (float*)d_out;

  u16* qT = (u16*)d_ws;                              // [N][HW][C] bf16, 4 MB
  u16* kT = qT + (size_t)NBATCH * HWD * CCH;         // [N][HW][C] bf16, 4 MB
  u16* vP = kT + (size_t)NBATCH * HWD * CCH;         // [N][C][HW] bf16, 4 MB

  dim3 gProj(HWD / 64, NBATCH, 3);
  proj_kernel<<<gProj, 128, 0, stream>>>(q, k, v, Wq, bq, Wk, bk, Wv, bv, qT, kT, vP);

  dim3 gAttn(HWD / 64, NBATCH);
  attn_kernel<<<gAttn, 128, 0, stream>>>(q, qT, kT, vP, out);

  (void)in_sizes; (void)n_in; (void)out_size; (void)ws_size;
}
